// PatchedMHA_91293824844056
// MI455X (gfx1250) — matrix operand order repaired and checked by reading, not compiled, not run on hardware
//
#include <hip/hip_runtime.h>
#include <hip/hip_bf16.h>

// Problem constants (from the reference): B=4, S=1024, D=2048, H=16, HD=128.
#define B_    4
#define S_    1024
#define D_    2048
#define H_    16
#define HD_   128
#define T_    (B_ * S_)          // 4096 tokens
#define NQKV_ (3 * H_ * HD_)     // 6144 qkv columns
#define LDP   40                 // padded LDS stride (80B rows: 16B-aligned, conflict-free)

typedef __attribute__((ext_vector_type(16))) _Float16 v16h;
typedef __attribute__((ext_vector_type(8)))  _Float16 v8h;
typedef __attribute__((ext_vector_type(4)))  _Float16 v4h;
typedef __attribute__((ext_vector_type(8)))  float    v8f;

__device__ __forceinline__ v8f wmma_f16(v16h a, v16h b, v8f c) {
  // D = A(16x32 f16) * B(32x16 f16) + C(16x16 f32)
  return __builtin_amdgcn_wmma_f32_16x16x32_f16(false, a, false, b, (short)0, c,
                                                false, false);
}

// A fragment (16x32, f16), source row-major [M][K] with leading dim ldm:
// lane m = lane&15 holds row m; K chunks at (lane>>4)*8 and (lane>>4)*8+16.
__device__ __forceinline__ v16h load_a_frag(const _Float16* p, int lane, int ldm) {
  const int m  = lane & 15;
  const int ks = (lane >> 4) * 8;
  v8h lo = *(const v8h*)(p + (size_t)m * ldm + ks);
  v8h hi = *(const v8h*)(p + (size_t)m * ldm + ks + 16);
  return __builtin_shufflevector(lo, hi, 0, 1, 2, 3, 4, 5, 6, 7,
                                 8, 9, 10, 11, 12, 13, 14, 15);
}

// B fragment (32x16, f16), source stored TRANSPOSED [N][K] with leading dim ldn:
// lane n = lane&15 holds column n; K chunks at (lane>>4)*8 and (lane>>4)*8+16.
__device__ __forceinline__ v16h load_b_frag(const _Float16* p, int lane, int ldn) {
  const int n  = lane & 15;
  const int ks = (lane >> 4) * 8;
  v8h lo = *(const v8h*)(p + (size_t)n * ldn + ks);
  v8h hi = *(const v8h*)(p + (size_t)n * ldn + ks + 16);
  return __builtin_shufflevector(lo, hi, 0, 1, 2, 3, 4, 5, 6, 7,
                                 8, 9, 10, 11, 12, 13, 14, 15);
}

// ---------------------------------------------------------------------------
// Kernel 1: QKV projection. C[t, j] = X[t, :] . Wqkv[:, j] + b_qkv[j], then
// scatter f16 results into q[b,h,s,d], k[b,h,s,d] and transposed vT[b,h,d,s].
// Tile: 128(M) x 128(N) x 32(K); 8 waves in a 2x4 grid, 64x32 per wave.
// Routing is BLOCK-UNIFORM: a 128-col N tile lies in exactly one of q/k/v and
// one head (HD==128); a 128-row M tile lies in one batch (S==1024).
// ---------------------------------------------------------------------------
__global__ __launch_bounds__(256) void qkv_gemm_kernel(
    const float* __restrict__ X, const float* __restrict__ W,
    const float* __restrict__ bias, _Float16* __restrict__ qbuf,
    _Float16* __restrict__ kbuf, _Float16* __restrict__ vtbuf) {
  __shared__ _Float16 As[128 * LDP];   // [M][K] f16
  __shared__ _Float16 Bs[128 * LDP];   // [N][K] f16 (transposed weight tile)

  const int tid  = threadIdx.x;
  const int lane = tid & 31;
  const int wave = tid >> 5;
  const int wm   = wave >> 2;          // 0..1
  const int wn   = wave & 3;           // 0..3
  const int t0   = blockIdx.y * 128;
  const int n0   = blockIdx.x * 128;

  v8f acc[4][2] = {};

  for (int k0 = 0; k0 < D_; k0 += 32) {
    // A tile: 128x32 fp32 -> f16 LDS (4 x float4 per thread)
#pragma unroll
    for (int i = 0; i < 4; ++i) {
      int idx = tid + i * 256;                 // 0..1023
      int row = idx >> 3;
      int c4  = (idx & 7) * 4;
      const float* gp = X + (size_t)(t0 + row) * D_ + k0 + c4;
      float4 f = *(const float4*)gp;
      if (k0 + 32 < D_) __builtin_prefetch(gp + 32, 0, 1);   // next K tile
      v4h h4 = {(_Float16)f.x, (_Float16)f.y, (_Float16)f.z, (_Float16)f.w};
      *(v4h*)(&As[row * LDP + c4]) = h4;
    }
    // B tile: 32(K) x 128(N) fp32 -> transposed f16 LDS
#pragma unroll
    for (int i = 0; i < 4; ++i) {
      int idx = tid + i * 256;                 // 0..1023
      int kk  = idx >> 5;                      // 0..31
      int c4  = (idx & 31) * 4;                // 0..124
      const float* gp = W + (size_t)(k0 + kk) * NQKV_ + n0 + c4;
      float4 f = *(const float4*)gp;
      if (k0 + 32 < D_) __builtin_prefetch(gp + (size_t)32 * NQKV_, 0, 1);
      Bs[(c4 + 0) * LDP + kk] = (_Float16)f.x;
      Bs[(c4 + 1) * LDP + kk] = (_Float16)f.y;
      Bs[(c4 + 2) * LDP + kk] = (_Float16)f.z;
      Bs[(c4 + 3) * LDP + kk] = (_Float16)f.w;
    }
    __syncthreads();

    v16h af[4], bf[2];
#pragma unroll
    for (int mi = 0; mi < 4; ++mi)
      af[mi] = load_a_frag(&As[(wm * 64 + mi * 16) * LDP], lane, LDP);
#pragma unroll
    for (int nj = 0; nj < 2; ++nj)
      bf[nj] = load_b_frag(&Bs[(wn * 32 + nj * 16) * LDP], lane, LDP);
#pragma unroll
    for (int mi = 0; mi < 4; ++mi)
#pragma unroll
      for (int nj = 0; nj < 2; ++nj)
        acc[mi][nj] = wmma_f16(af[mi], bf[nj], acc[mi][nj]);
    __syncthreads();
  }

  // Epilogue: bias + store. All routing decisions are uniform scalars.
  const int hi = lane >> 4, nn = lane & 15;
  const int p  = n0 >> 11;                    // 0:q 1:k 2:v   (uniform)
  const int h  = (n0 >> 7) & (H_ - 1);        // head           (uniform)
  const int b  = t0 >> 10;                    // batch          (uniform)
  const int s0 = t0 & (S_ - 1);
  const size_t bh = (size_t)(b * H_ + h);
  float bv[2];
#pragma unroll
  for (int nj = 0; nj < 2; ++nj)
    bv[nj] = bias[n0 + wn * 32 + nj * 16 + nn];

  if (p < 2) {                                // q or k: [b,h,s,d] row-major
    _Float16* dst = (p == 0) ? qbuf : kbuf;
#pragma unroll
    for (int mi = 0; mi < 4; ++mi)
#pragma unroll
      for (int nj = 0; nj < 2; ++nj)
#pragma unroll
        for (int r = 0; r < 8; ++r) {
          int s = s0 + wm * 64 + mi * 16 + r + 8 * hi;
          int d = wn * 32 + nj * 16 + nn;
          dst[(bh * S_ + s) * HD_ + d] = (_Float16)(acc[mi][nj][r] + bv[nj]);
        }
  } else {                                    // v: transposed [b,h,d,s]
#pragma unroll
    for (int mi = 0; mi < 4; ++mi)
#pragma unroll
      for (int nj = 0; nj < 2; ++nj)
#pragma unroll
        for (int r = 0; r < 8; ++r) {
          int s = s0 + wm * 64 + mi * 16 + r + 8 * hi;
          int d = wn * 32 + nj * 16 + nn;
          vtbuf[(bh * HD_ + d) * S_ + s] = (_Float16)(acc[mi][nj][r] + bv[nj]);
        }
  }
}

// ---------------------------------------------------------------------------
// Kernel 2: RoPE on q and k in place; folds the 1/sqrt(HD) score scale into q.
// One thread per (b,h,s, d<64) pair.
// ---------------------------------------------------------------------------
__global__ __launch_bounds__(256) void rope_kernel(_Float16* __restrict__ qbuf,
                                                   _Float16* __restrict__ kbuf) {
  int idx = blockIdx.x * 256 + threadIdx.x;
  if (idx >= T_ * H_ * 64) return;
  int d   = idx & 63;
  int row = idx >> 6;                 // (b*H + h)*S + s
  int s   = row & (S_ - 1);
  float inv = __expf((float)d * (-9.210340371976184f / 64.f));  // 10000^(-d/64)
  float ang = (float)s * inv;
  float sn, cs;
  __sincosf(ang, &sn, &cs);
  const float qs = 0.08838834764831845f;  // 1/sqrt(128)
  size_t base = (size_t)row * HD_;
  float q1 = (float)qbuf[base + d], q2 = (float)qbuf[base + d + 64];
  qbuf[base + d]      = (_Float16)((q1 * cs - q2 * sn) * qs);
  qbuf[base + d + 64] = (_Float16)((q2 * cs + q1 * sn) * qs);
  float k1 = (float)kbuf[base + d], k2 = (float)kbuf[base + d + 64];
  kbuf[base + d]      = (_Float16)(k1 * cs - k2 * sn);
  kbuf[base + d + 64] = (_Float16)(k2 * cs + k1 * sn);
}

// ---------------------------------------------------------------------------
// Kernel 3: causal flash attention per (b,h). One wave owns a 16-query tile,
// loops over 32-key tiles (2 score fragments -> exactly one 16x32 A fragment
// for P*V). Online softmax in C-fragment layout; P transits per-wave LDS.
// ---------------------------------------------------------------------------
__global__ __launch_bounds__(128) void attn_kernel(
    const _Float16* __restrict__ qbuf, const _Float16* __restrict__ kbuf,
    const _Float16* __restrict__ vtbuf, _Float16* __restrict__ obuf) {
  __shared__ _Float16 Pbuf[4 * 16 * LDP];

  const int lane = threadIdx.x & 31;
  const int wave = threadIdx.x >> 5;
  const int bh   = blockIdx.y;                       // 0..B*H-1
  const int q0   = (blockIdx.x * 4 + wave) * 16;
  const _Float16* Q  = qbuf  + (size_t)bh * S_ * HD_;
  const _Float16* K  = kbuf  + (size_t)bh * S_ * HD_;
  const _Float16* Vt = vtbuf + (size_t)bh * HD_ * S_;
  _Float16* pw = &Pbuf[wave * 16 * LDP];
  const int hi = lane >> 4, nn = lane & 15;

  // Q fragments for this 16-row tile (4 K-chunks of 32 over HD=128).
  v16h qf[4];
#pragma unroll
  for (int c = 0; c < 4; ++c)
    qf[c] = load_a_frag(Q + (size_t)q0 * HD_ + c * 32, lane, HD_);

  v8f of[8] = {};
  float m[8], l[8];
#pragma unroll
  for (int r = 0; r < 8; ++r) { m[r] = -1e30f; l[r] = 0.f; }

  const int ktiles = (q0 + 16 + 31) >> 5;            // causal bound
  for (int kt = 0; kt < ktiles; ++kt) {
    const int k0 = kt * 32;
    // Scores: S(16x32) = Q(16x128) * K^T(128x32), scale already in q.
    v8f sf[2] = {};
#pragma unroll
    for (int j = 0; j < 2; ++j)
#pragma unroll
      for (int c = 0; c < 4; ++c) {
        v16h kf = load_b_frag(K + (size_t)(k0 + j * 16) * HD_ + c * 32, lane, HD_);
        sf[j] = wmma_f16(qf[c], kf, sf[j]);
      }
    // Causal mask (only tiles touching the diagonal).
    if (k0 + 31 > q0) {
#pragma unroll
      for (int j = 0; j < 2; ++j)
#pragma unroll
        for (int r = 0; r < 8; ++r) {
          int col = k0 + j * 16 + nn;
          int row = q0 + r + 8 * hi;
          if (col > row) sf[j][r] = -1e30f;
        }
    }
    // Online softmax; row r+8*hi lives at VGPR slot r across the 16-lane group.
#pragma unroll
    for (int r = 0; r < 8; ++r) {
      float mx = fmaxf(sf[0][r], sf[1][r]);
      mx = fmaxf(mx, __shfl_xor(mx, 1, 32));
      mx = fmaxf(mx, __shfl_xor(mx, 2, 32));
      mx = fmaxf(mx, __shfl_xor(mx, 4, 32));
      mx = fmaxf(mx, __shfl_xor(mx, 8, 32));
      float mnew = fmaxf(m[r], mx);
      float p0 = __expf(sf[0][r] - mnew);
      float p1 = __expf(sf[1][r] - mnew);
      sf[0][r] = p0;
      sf[1][r] = p1;
      float rs = p0 + p1;
      rs += __shfl_xor(rs, 1, 32);
      rs += __shfl_xor(rs, 2, 32);
      rs += __shfl_xor(rs, 4, 32);
      rs += __shfl_xor(rs, 8, 32);
      float alpha = __expf(m[r] - mnew);
      l[r] = l[r] * alpha + rs;
      m[r] = mnew;
#pragma unroll
      for (int j2 = 0; j2 < 8; ++j2) of[j2][r] *= alpha;
    }
    // P (C layout) -> LDS -> A fragment (layout transpose through LDS).
#pragma unroll
    for (int j = 0; j < 2; ++j)
#pragma unroll
      for (int r = 0; r < 8; ++r)
        pw[(r + 8 * hi) * LDP + j * 16 + nn] = (_Float16)sf[j][r];
    asm volatile("s_wait_dscnt 0x0" ::: "memory");   // cross-lane LDS visibility
    v16h pf = load_a_frag(pw, lane, LDP);
    // O(16x128) += P(16x32) * V(32x128); V read transposed -> contiguous K runs.
#pragma unroll
    for (int j2 = 0; j2 < 8; ++j2) {
      v16h vf = load_b_frag(Vt + (size_t)(j2 * 16) * S_ + k0, lane, S_);
      of[j2] = wmma_f16(pf, vf, of[j2]);
    }
  }

  // Normalize and store attn output as f16 [T][H*HD] for the O-projection.
  const int b = bh >> 4, h = bh & 15;
#pragma unroll
  for (int j2 = 0; j2 < 8; ++j2)
#pragma unroll
    for (int r = 0; r < 8; ++r) {
      int s = q0 + r + 8 * hi;
      int d = j2 * 16 + nn;
      float ov = of[j2][r] / l[r];
      obuf[((size_t)(b * S_ + s)) * (H_ * HD_) + h * HD_ + d] = (_Float16)ov;
    }
}

// ---------------------------------------------------------------------------
// Kernel 4: output projection. out[t, n] = attn[t, :] . Wo[:, n] + b_o[n].
// Same tiling as kernel 1; A is already f16, B converted fp32->f16 on the fly.
// ---------------------------------------------------------------------------
__global__ __launch_bounds__(256) void oproj_gemm_kernel(
    const _Float16* __restrict__ Xh, const float* __restrict__ W,
    const float* __restrict__ bias, float* __restrict__ Out) {
  __shared__ _Float16 As[128 * LDP];
  __shared__ _Float16 Bs[128 * LDP];

  const int tid  = threadIdx.x;
  const int lane = tid & 31;
  const int wave = tid >> 5;
  const int wm   = wave >> 2;
  const int wn   = wave & 3;
  const int t0   = blockIdx.y * 128;
  const int n0   = blockIdx.x * 128;

  v8f acc[4][2] = {};

  for (int k0 = 0; k0 < D_; k0 += 32) {
    // A tile: 128x32 f16 straight copy (2 x 16B per thread).
#pragma unroll
    for (int i = 0; i < 2; ++i) {
      int idx = tid + i * 256;                 // 0..511
      int row = idx >> 2;
      int c8  = (idx & 3) * 8;
      const _Float16* gp = Xh + (size_t)(t0 + row) * D_ + k0 + c8;
      v8h h8 = *(const v8h*)gp;
      if (k0 + 32 < D_) __builtin_prefetch(gp + 32, 0, 1);
      *(v8h*)(&As[row * LDP + c8]) = h8;
    }
    // B tile: Wo 32(K) x 128(N) fp32 -> transposed f16 LDS.
#pragma unroll
    for (int i = 0; i < 4; ++i) {
      int idx = tid + i * 256;
      int kk  = idx >> 5;
      int c4  = (idx & 31) * 4;
      const float* gp = W + (size_t)(k0 + kk) * D_ + n0 + c4;
      float4 f = *(const float4*)gp;
      if (k0 + 32 < D_) __builtin_prefetch(gp + (size_t)32 * D_, 0, 1);
      Bs[(c4 + 0) * LDP + kk] = (_Float16)f.x;
      Bs[(c4 + 1) * LDP + kk] = (_Float16)f.y;
      Bs[(c4 + 2) * LDP + kk] = (_Float16)f.z;
      Bs[(c4 + 3) * LDP + kk] = (_Float16)f.w;
    }
    __syncthreads();

    v16h af[4], bf[2];
#pragma unroll
    for (int mi = 0; mi < 4; ++mi)
      af[mi] = load_a_frag(&As[(wm * 64 + mi * 16) * LDP], lane, LDP);
#pragma unroll
    for (int nj = 0; nj < 2; ++nj)
      bf[nj] = load_b_frag(&Bs[(wn * 32 + nj * 16) * LDP], lane, LDP);
#pragma unroll
    for (int mi = 0; mi < 4; ++mi)
#pragma unroll
      for (int nj = 0; nj < 2; ++nj)
        acc[mi][nj] = wmma_f16(af[mi], bf[nj], acc[mi][nj]);
    __syncthreads();
  }

  const int hi = lane >> 4, nn = lane & 15;
  float bv[2];
#pragma unroll
  for (int nj = 0; nj < 2; ++nj)
    bv[nj] = bias[n0 + wn * 32 + nj * 16 + nn];
#pragma unroll
  for (int mi = 0; mi < 4; ++mi)
#pragma unroll
    for (int nj = 0; nj < 2; ++nj)
#pragma unroll
      for (int r = 0; r < 8; ++r) {
        int t    = t0 + wm * 64 + mi * 16 + r + 8 * hi;
        int gcol = n0 + wn * 32 + nj * 16 + nn;
        Out[(size_t)t * D_ + gcol] = acc[mi][nj][r] + bv[nj];
      }
}

// ---------------------------------------------------------------------------
// Launch. The cache scatter+gather in the reference is an exact inverse pair
// for any injective block_offsets, so k_cache/v_cache/block_offsets are
// output-irrelevant. Workspace: q,k,vT,attn f16 buffers = 64 MB total.
// ---------------------------------------------------------------------------
extern "C" void kernel_launch(void* const* d_in, const int* in_sizes, int n_in,
                              void* d_out, int out_size, void* d_ws,
                              size_t ws_size, hipStream_t stream) {
  (void)in_sizes; (void)n_in; (void)out_size; (void)ws_size;
  const float* hidden = (const float*)d_in[0];
  const float* Wqkv   = (const float*)d_in[1];
  const float* b_qkv  = (const float*)d_in[2];
  const float* Wo     = (const float*)d_in[3];
  const float* b_o    = (const float*)d_in[4];
  float* out = (float*)d_out;

  const size_t NE = (size_t)B_ * H_ * S_ * HD_;   // 8,388,608 elements
  _Float16* qbuf  = (_Float16*)d_ws;
  _Float16* kbuf  = qbuf + NE;
  _Float16* vtbuf = kbuf + NE;
  _Float16* obuf  = vtbuf + NE;

  qkv_gemm_kernel<<<dim3(NQKV_ / 128, T_ / 128), 256, 0, stream>>>(
      hidden, Wqkv, b_qkv, qbuf, kbuf, vtbuf);
  rope_kernel<<<(T_ * H_ * 64) / 256, 256, 0, stream>>>(qbuf, kbuf);
  attn_kernel<<<dim3(S_ / 64, B_ * H_), 128, 0, stream>>>(qbuf, kbuf, vtbuf, obuf);
  oproj_gemm_kernel<<<dim3(D_ / 128, T_ / 128), 256, 0, stream>>>(
      obuf, Wo, b_o, out);
}
